// MixtureOfExperts_34050500723197
// MI455X (gfx1250) — matrix operand order repaired and checked by reading, not compiled, not run on hardware
//
#include <hip/hip_runtime.h>
#include <hip/hip_bf16.h>

// ---------------- types ----------------
typedef __attribute__((ext_vector_type(16))) __bf16    v16bf;
typedef __attribute__((ext_vector_type(8)))  float     v8f;
typedef __attribute__((ext_vector_type(4)))  unsigned  u32x4;
typedef __attribute__((ext_vector_type(8)))  int       i32x8;
typedef __attribute__((ext_vector_type(4)))  int       i32x4;
typedef __attribute__((ext_vector_type(4)))  float     f32x4;
typedef __attribute__((ext_vector_type(2)))  float     f32x2;

// ---------------- problem sizes ----------------
#define BATCH 8192
#define NEXP  64
#define NCAP  256
#define DIN   16384          // NEXP*NCAP
#define HID1  256
#define HID2  128
#define TOPK  8

#define NT    (HID1 / 16)    // 16 n-tiles of 16 cols
#define NS    (DIN / 32)     // 512 k-steps of 32
#define BCHUNK_DW (2 * NT * 32 * 8)   // 8192 dwords staged per 64-wide k-chunk

#if __has_builtin(__builtin_amdgcn_tensor_load_to_lds)
#define HAVE_TDM 1
#else
#define HAVE_TDM 0
#endif

// round-to-nearest-even fp32 -> bf16, packed pair into one dword
__device__ __forceinline__ unsigned pk2bf(float a, float b) {
  unsigned ua = __float_as_uint(a); ua += 0x7FFFu + ((ua >> 16) & 1u);
  unsigned ub = __float_as_uint(b); ub += 0x7FFFu + ((ub >> 16) & 1u);
  return (ua >> 16) | (ub & 0xFFFF0000u);
}

#if HAVE_TDM
// 1D linear copy of `ndw` dwords from global `gsrc` to LDS byte offset `lds_off`
// via the Tensor Data Mover.  D# per CDNA5 ISA ch.8:
//   group0: [1:0]=count=1, [63:32]=lds_addr, [120:64]=global_addr, [127:126]=type=2
//   group1: [17:16]=data_size(2 -> 4B), [79:48]=tensor_dim0, [111:80]=tensor_dim1=1,
//           [127:112]=tile_dim0, [143:128]=tile_dim1=1, [207:160]=tensor_dim0_stride
__device__ __forceinline__ void tdm_load_lds(const unsigned* gsrc, unsigned lds_off,
                                             unsigned ndw) {
  unsigned long long ga = (unsigned long long)gsrc;
  u32x4 g0;
  g0.x = 1u;                                    // count=1, user descriptor
  g0.y = lds_off;                               // LDS byte address
  g0.z = (unsigned)ga;                          // global_addr[31:0]
  g0.w = (unsigned)(ga >> 32) | (2u << 30);     // global_addr[56:32] | type=2
  i32x8 g1;
  g1[0] = (int)(2u << 16);                      // data_size code 2 = 4 bytes
  g1[1] = (int)(ndw << 16);                     // tensor_dim0[15:0] @ bits 63:48
  g1[2] = (int)((ndw >> 16) | (1u << 16));      // tensor_dim0[31:16] | tensor_dim1=1
  g1[3] = (int)(ndw << 16);                     // tile_dim0 @ bits 127:112
  g1[4] = 1;                                    // tile_dim1=1, tile_dim2=0
  g1[5] = (int)ndw;                             // tensor_dim0_stride[31:0]
  g1[6] = 0;
  g1[7] = 0;
  i32x4 g2 = {0, 0, 0, 0};
  i32x4 g3 = {0, 0, 0, 0};
#if defined(__clang_major__) && (__clang_major__ >= 23)
  i32x8 g4 = {0, 0, 0, 0, 0, 0, 0, 0};
  __builtin_amdgcn_tensor_load_to_lds(g0, g1, g2, g3, g4, 0);
#else
  __builtin_amdgcn_tensor_load_to_lds(g0, g1, g2, g3, 0);
#endif
}
#endif

// ============================================================
// Kernel 0: repack W1 (DIN x HID1, fp32 row-major) into bf16
// WMMA B-fragment layout:
//   dword index = ((s*NT + t)*32 + lane)*8 + v
//   lane<16: col = t*16+lane, K = s*32 + 2v       (pair K,K+1)
//   lane>=16: col = t*16+(lane-16), K = s*32+16+2v
// ============================================================
__global__ __launch_bounds__(256) void pack_w1(const float* __restrict__ W1,
                                               unsigned* __restrict__ W1f) {
  unsigned gid  = blockIdx.x * 256u + threadIdx.x;   // 0 .. 2M-1 dwords
  unsigned v    = gid & 7u;
  unsigned lane = (gid >> 3) & 31u;
  unsigned t    = (gid >> 8) & 15u;
  unsigned s    = gid >> 12;
  unsigned half = lane >> 4, nl = lane & 15u;
  unsigned n = t * 16u + nl;
  unsigned k = s * 32u + 16u * half + 2u * v;
  W1f[gid] = pk2bf(W1[(size_t)k * HID1 + n], W1[(size_t)(k + 1) * HID1 + n]);
}

// ============================================================
// Kernel 1: h1 = relu(A @ W1 + b1) via v_wmma_f32_16x16x32_bf16
//   block: 256 thr = 8 waves, tile 32 rows x 256 cols, Kchunk 64
//   B fragments DMA'd to LDS by the Tensor Data Mover.
// ============================================================
__global__ __launch_bounds__(256) void gemm1(const float* __restrict__ A,
                                             const unsigned* __restrict__ Bf,
                                             const float* __restrict__ b1,
                                             float* __restrict__ h1out) {
  __shared__ __align__(16) unsigned short Als[32 * 72];        // 32 rows x 64 bf16 (+8 pad)
  __shared__ __align__(16) unsigned       Bls[BCHUNK_DW];      // 8192 dwords = 32 KB

  const int tid  = threadIdx.x;
  const int wave = tid >> 5, lane = tid & 31;
  const int half = lane >> 4, nl = lane & 15;
  const int mt   = wave >> 2;            // 0..1: which 16-row tile
  const int ng   = (wave & 3) * 4;       // first of 4 owned n-tiles
  const size_t m0 = (size_t)blockIdx.x * 32;

  const int arow = tid >> 3;             // 0..31
  const int acb  = (tid & 7) * 8;        // 0,8,...,56
  const float* aptr = A + m0 * DIN + (size_t)arow * DIN + acb;

#if HAVE_TDM
  const unsigned bls_off = (unsigned)(unsigned long long)(void*)Bls;
#endif

  v8f acc[4] = {};

  for (int ks = 0; ks < DIN; ks += 64) {
    // prefetch next A chunk (speculative; wraps on last iter)
    __builtin_prefetch(aptr + ((ks + 64) & (DIN - 1)), 0, 0);

    // --- stage B fragments: 8192 dwords (32 KB), linear ---
#if HAVE_TDM
    if (wave == 0) {
      tdm_load_lds(Bf + (size_t)(ks >> 5) * (NT * 32 * 8), bls_off, BCHUNK_DW);
    }
#else
    {
      const u32x4* src = (const u32x4*)(Bf + (size_t)(ks >> 5) * (NT * 32 * 8));
      u32x4* dst = (u32x4*)Bls;
#pragma unroll
      for (int i = 0; i < 8; ++i) dst[tid + i * 256] = src[tid + i * 256];
    }
#endif

    // --- stage A chunk: 32x64 fp32 -> bf16 in LDS ---
    {
      const f32x4* p = (const f32x4*)(aptr + ks);
      f32x4 f0 = p[0], f1 = p[1];
      u32x4 dw;
      dw.x = pk2bf(f0.x, f0.y); dw.y = pk2bf(f0.z, f0.w);
      dw.z = pk2bf(f1.x, f1.y); dw.w = pk2bf(f1.z, f1.w);
      *(u32x4*)&Als[arow * 72 + acb] = dw;
    }

#if HAVE_TDM
    if (wave == 0) __builtin_amdgcn_s_wait_tensorcnt(0);
#endif
    __syncthreads();

#pragma unroll
    for (int sub = 0; sub < 2; ++sub) {
      // A fragment (16x32 bf16): lanes0-15 K 0..15 ; lanes16-31 K 16..31
      union { u32x4 q[2]; v16bf v; } af;
      const int row = mt * 16 + nl;
      const int kb  = sub * 32 + 16 * half;
      af.q[0] = *(const u32x4*)&Als[row * 72 + kb];
      af.q[1] = *(const u32x4*)&Als[row * 72 + kb + 8];
#pragma unroll
      for (int j = 0; j < 4; ++j) {
        union { u32x4 q[2]; v16bf v; } bfv;
        const unsigned* bp = &Bls[((sub * NT + (ng + j)) * 32 + lane) * 8];
        bfv.q[0] = *(const u32x4*)bp;
        bfv.q[1] = *(const u32x4*)(bp + 4);
        acc[j] = __builtin_amdgcn_wmma_f32_16x16x32_bf16(
            false, af.v, false, bfv.v, (short)0, acc[j], false, false);
      }
    }
    __syncthreads();
  }

  // epilogue: bias + relu, C/D layout: lanes0-15 M=vgpr, lanes16-31 M=vgpr+8, N=lane&15
#pragma unroll
  for (int j = 0; j < 4; ++j) {
    const int col  = (ng + j) * 16 + nl;
    const float bb = b1[col];
#pragma unroll
    for (int i = 0; i < 8; ++i) {
      const int mloc = mt * 16 + half * 8 + i;
      float vv = acc[j][i] + bb;
      h1out[(m0 + mloc) * HID1 + col] = vv > 0.f ? vv : 0.f;
    }
  }
}

// ============================================================
// Kernel 2: h2 = relu(h1@W2+b2); logits = h2@W3+b3; softmax;
//           top-8 + renormalize. One batch row per wave.
// ============================================================
__global__ __launch_bounds__(256) void gate(const float* __restrict__ h1,
                                            const float* __restrict__ W2,
                                            const float* __restrict__ b2,
                                            const float* __restrict__ W3,
                                            const float* __restrict__ b3,
                                            int* __restrict__ topIdx,
                                            float* __restrict__ topW) {
  __shared__ __align__(16) float h1s[8][HID1];
  __shared__ __align__(16) float h2s[8][HID2];
  const int tid = threadIdx.x, wave = tid >> 5, lane = tid & 31;
  const size_t b = (size_t)blockIdx.x * 8 + wave;

#pragma unroll
  for (int i = 0; i < 8; ++i)
    h1s[wave][lane + i * 32] = h1[b * HID1 + lane + i * 32];
  __syncthreads();

  // h2: each lane owns 4 consecutive cols
  float a0 = 0.f, a1 = 0.f, a2 = 0.f, a3 = 0.f;
  for (int i = 0; i < HID1; ++i) {
    float hv = h1s[wave][i];
    f32x4 w = *(const f32x4*)&W2[(size_t)i * HID2 + lane * 4];
    a0 += hv * w.x; a1 += hv * w.y; a2 += hv * w.z; a3 += hv * w.w;
  }
  {
    f32x4 bb = *(const f32x4*)&b2[lane * 4];
    h2s[wave][lane * 4 + 0] = fmaxf(a0 + bb.x, 0.f);
    h2s[wave][lane * 4 + 1] = fmaxf(a1 + bb.y, 0.f);
    h2s[wave][lane * 4 + 2] = fmaxf(a2 + bb.z, 0.f);
    h2s[wave][lane * 4 + 3] = fmaxf(a3 + bb.w, 0.f);
  }
  __syncthreads();

  // logits: each lane owns 2 consecutive experts
  float l0 = 0.f, l1 = 0.f;
  for (int i = 0; i < HID2; ++i) {
    float hv = h2s[wave][i];
    f32x2 w = *(const f32x2*)&W3[(size_t)i * NEXP + lane * 2];
    l0 += hv * w.x; l1 += hv * w.y;
  }
  l0 += b3[lane * 2]; l1 += b3[lane * 2 + 1];

  // softmax over 64 (wave32 butterfly reductions)
  float m = fmaxf(l0, l1);
  for (int off = 16; off; off >>= 1) m = fmaxf(m, __shfl_xor(m, off, 32));
  float e0 = __expf(l0 - m), e1 = __expf(l1 - m);
  float ssum = e0 + e1;
  for (int off = 16; off; off >>= 1) ssum += __shfl_xor(ssum, off, 32);
  float g0 = e0 / ssum, g1 = e1 / ssum;

  // top-8 via 8 argmax sweeps
  float myV = 0.f, tsum = 0.f;
  int myI = 0;
#pragma unroll
  for (int k = 0; k < TOPK; ++k) {
    float mv; int mi;
    if (g0 >= g1) { mv = g0; mi = lane * 2; } else { mv = g1; mi = lane * 2 + 1; }
    for (int off = 16; off; off >>= 1) {
      float ov = __shfl_xor(mv, off, 32);
      int   oi = __shfl_xor(mi, off, 32);
      if (ov > mv || (ov == mv && oi < mi)) { mv = ov; mi = oi; }
    }
    tsum += mv;
    if (lane == k) { myV = mv; myI = mi; }
    if (mi == lane * 2) g0 = -1.f;
    else if (mi == lane * 2 + 1) g1 = -1.f;
  }
  if (lane < TOPK) {
    topIdx[b * TOPK + lane] = myI;
    topW [b * TOPK + lane] = myV / tsum;
  }
}

// ============================================================
// Kernel 3: out[b][c] = sum_k w[b][k] * probs[b][idx[b][k]][c]
// ============================================================
__global__ __launch_bounds__(256) void final_mix(const float* __restrict__ probs,
                                                 const int* __restrict__ topIdx,
                                                 const float* __restrict__ topW,
                                                 float* __restrict__ out) {
  const size_t b = blockIdx.x;
  const int c = threadIdx.x;
  float acc = 0.f;
#pragma unroll
  for (int k = 0; k < TOPK; ++k) {
    int   e = topIdx[b * TOPK + k];
    float w = topW [b * TOPK + k];
    acc += w * probs[((size_t)b * NEXP + e) * NCAP + c];
  }
  out[b * NCAP + c] = acc;
}

// ============================================================
extern "C" void kernel_launch(void* const* d_in, const int* in_sizes, int n_in,
                              void* d_out, int out_size, void* d_ws, size_t ws_size,
                              hipStream_t stream) {
  const float* probs = (const float*)d_in[0];
  const float* W1    = (const float*)d_in[1];
  const float* b1    = (const float*)d_in[2];
  const float* W2    = (const float*)d_in[3];
  const float* b2    = (const float*)d_in[4];
  const float* W3    = (const float*)d_in[5];
  const float* b3    = (const float*)d_in[6];
  float* out = (float*)d_out;

  char* ws = (char*)d_ws;
  unsigned* W1f  = (unsigned*)ws;                                  // 8 MB
  float*    h1   = (float*)(ws + (size_t)8 * 1024 * 1024);         // 8 MB
  int*      tIdx = (int*)  (ws + (size_t)16 * 1024 * 1024);        // 256 KB
  float*    tW   = (float*)(ws + (size_t)16 * 1024 * 1024 + 256 * 1024);

  pack_w1 <<<(DIN * HID1 / 2) / 256, 256, 0, stream>>>(W1, W1f);
  gemm1   <<<BATCH / 32,            256, 0, stream>>>(probs, W1f, b1, h1);
  gate    <<<BATCH / 8,             256, 0, stream>>>(h1, W2, b2, W3, b3, tIdx, tW);
  final_mix<<<BATCH,                256, 0, stream>>>(probs, tIdx, tW, out);
}
